// MoEHeadAdapter_30502857736781
// MI455X (gfx1250) — matrix operand order repaired and checked by reading, not compiled, not run on hardware
//
#include <hip/hip_runtime.h>
#include <hip/hip_bf16.h>

// MoE head adapter, fused for MI455X (gfx1250, wave32, WMMA, async-to-LDS).
// Pipeline: [pack f32->f16 fragment-ordered weights] -> [top-2 gate kernel]
//   -> fused kernel: double-buffered LDS weight stages filled by
//      global_load_async_to_lds_b128 one stage ahead of the WMMA consumer.
// Stage schedule (all 16KB contiguous chunks of packed workspace):
//   loop A (64x): [GEMM1 w1(e,c2) from buf0 | prefetch w2 -> buf1]
//                 [GEMM2 w2(e,c2) from buf1 | prefetch next w1 -> buf0]
//   loop B (16x): GEMM3 wp(t) ping-pong.
// Workspace (dwords): w1p[262144] | w2p[262144] | wpp[65536] | gates[262144]
// => needs ~3.26 MB of d_ws.

typedef __attribute__((ext_vector_type(16))) _Float16     v16h;
typedef __attribute__((ext_vector_type(8)))  float        v8f;
typedef __attribute__((ext_vector_type(8)))  unsigned int v8u;
typedef __attribute__((ext_vector_type(4)))  unsigned int v4u;
typedef __attribute__((ext_vector_type(4)))  float        f32x4;

#define WG_THREADS 256
#define WG_WAVES   8
#define D_IN    256
#define E_EXP   4
#define H_DIM   512
#define EMB_DIM 512

__device__ __forceinline__ v16h frag_from_dwords(const unsigned int* p) {
  v4u a = *(const v4u*)p;
  v4u b = *(const v4u*)(p + 4);
  v8u u = __builtin_shufflevector(a, b, 0, 1, 2, 3, 4, 5, 6, 7);
  return __builtin_bit_cast(v16h, u);
}

__device__ __forceinline__ v16h frag_from_halves(const _Float16* lo, const _Float16* hi16) {
  v4u a = *(const v4u*)lo;     // K+0..7   (8 x f16 = 16B)
  v4u b = *(const v4u*)hi16;   // K+8..15
  v8u u = __builtin_shufflevector(a, b, 0, 1, 2, 3, 4, 5, 6, 7);
  return __builtin_bit_cast(v16h, u);
}

__device__ __forceinline__ v8f wmma16(v16h a, v16h b, v8f c) {
  // D = A(16x32 f16) x B(32x16 f16) + C(16x16 f32)
  return __builtin_amdgcn_wmma_f32_16x16x32_f16(false, a, false, b, (short)0, c, false, false);
}

// --- CDNA5 async global->LDS copy (ASYNCcnt-tracked, bypasses VGPRs) -------
__device__ __forceinline__ void async_copy_b128(unsigned lds_byte_addr, const void* gaddr) {
  asm volatile("global_load_async_to_lds_b128 %0, %1, off"
               :
               : "v"(lds_byte_addr), "v"((unsigned long long)(uintptr_t)gaddr)
               : "memory");
}
__device__ __forceinline__ void wait_async0() {
  asm volatile("s_wait_asynccnt 0x0" ::: "memory");
}

// copy one 16KB stage chunk: 256 threads x 4 b128
__device__ __forceinline__ void stage_prefetch(const unsigned int* gbase_tid,
                                               unsigned lds_base_tid,
                                               unsigned src_dw_off) {
  const unsigned int* gp = gbase_tid + src_dw_off;
#pragma unroll
  for (int q = 0; q < 4; ++q) async_copy_b128(lds_base_tid + q * 16, gp + q * 4);
}

// ---------------------------------------------------------------------------
// Pack a set of B matrices [nE][Kdim=nC*32][Ndim] (f32, row-major) into
// f16 WMMA B-fragment blocks of 1KB: block = (e, j, c) [cj_order=0] or
// (e, c, j) [cj_order=1]; dword within block = lane*8 + v, value = packed
// f16 pair (K = base+2v, base+2v+1); lanes 0-15: col j*16+lane, K base c*32;
// lanes 16-31: col j*16+lane-16, K base c*32+16.  (contiguous K order, same as the A fragments)
// ---------------------------------------------------------------------------
__global__ __launch_bounds__(256) void pack_w_kernel(
    const float* __restrict__ src, unsigned int* __restrict__ dst,
    int nJ, int nC, int Ndim, int cj_order, int total)
{
  const int g = blockIdx.x * 256 + threadIdx.x;
  if (g >= total) return;
  const int v    = g & 7;
  const int lane = (g >> 3) & 31;
  const int bi0  = g >> 8;
  const int bpe  = nJ * nC;
  const int e    = bi0 / bpe;
  const int bi   = bi0 % bpe;
  int j, c;
  if (cj_order) { c = bi / nJ; j = bi % nJ; }
  else          { j = bi / nC; c = bi % nC; }
  const int col  = j * 16 + (lane & 15);
  const int k0   = c * 32 + (lane >> 4) * 16 + 2 * v;
  const int Kdim = nC * 32;
  const float* s = src + (size_t)e * Kdim * Ndim;
  _Float16 h0 = (_Float16)s[(size_t)k0 * Ndim + col];
  _Float16 h1 = (_Float16)s[(size_t)(k0 + 1) * Ndim + col];
  unsigned short u0, u1;
  __builtin_memcpy(&u0, &h0, 2);
  __builtin_memcpy(&u1, &h1, 2);
  dst[g] = (unsigned int)u0 | ((unsigned int)u1 << 16);
}

// ---------------------------------------------------------------------------
// Top-2 gating: one wave per token; exact fp32; dense [N,4] gate matrix.
// ---------------------------------------------------------------------------
__global__ __launch_bounds__(256) void gate_topk_kernel(
    const float* __restrict__ x, const float* __restrict__ w_gate,
    float* __restrict__ gates)
{
  const int n    = blockIdx.x * 8 + (threadIdx.x >> 5);
  const int lane = threadIdx.x & 31;
  const float* xr = x + (size_t)n * D_IN;
  float p0 = 0.f, p1 = 0.f, p2 = 0.f, p3 = 0.f;
#pragma unroll
  for (int it = 0; it < 8; ++it) {
    const int d = lane + it * 32;
    const float xv = xr[d];
    f32x4 wv = *(const f32x4*)(w_gate + d * 4);
    p0 += xv * wv[0]; p1 += xv * wv[1]; p2 += xv * wv[2]; p3 += xv * wv[3];
  }
#pragma unroll
  for (int off = 16; off > 0; off >>= 1) {
    p0 += __shfl_xor(p0, off, 32);
    p1 += __shfl_xor(p1, off, 32);
    p2 += __shfl_xor(p2, off, 32);
    p3 += __shfl_xor(p3, off, 32);
  }
  if (lane == 0) {
    float l[4] = {p0, p1, p2, p3};
    int i0 = 0;
#pragma unroll
    for (int e2 = 1; e2 < 4; ++e2) if (l[e2] > l[i0]) i0 = e2;
    int i1 = (i0 == 0) ? 1 : 0;
#pragma unroll
    for (int e2 = 0; e2 < 4; ++e2) if (e2 != i0 && l[e2] > l[i1]) i1 = e2;
    const float ex1 = expf(l[i1] - l[i0]);
    const float inv = 1.0f / (1.0f + ex1);
    f32x4 o = {0.f, 0.f, 0.f, 0.f};
    o[i0] = inv;
    o[i1] = ex1 * inv;
    *(f32x4*)(gates + (size_t)n * 4) = o;
  }
}

// ---------------------------------------------------------------------------
// Fused MoE kernel: 8 waves / WG, 16 tokens per wave (128 per WG).
// ---------------------------------------------------------------------------
__global__ __launch_bounds__(WG_THREADS) void moe_fused_kernel(
    const float* __restrict__ x,
    const float* __restrict__ b_proj,
    const unsigned int* __restrict__ wsall,   // packed weights base (dwords)
    const float* __restrict__ gates,
    float* __restrict__ out)
{
  __shared__ __align__(16) unsigned int s_stage[2 * 4096];    // 2 x 16 KB stage
  __shared__ __align__(16) _Float16 s_hb[WG_WAVES][16][32];   // 1 KB/wave bounce
  __shared__ float s_g[128 * 4];                              // gate matrix tile

  const int tid  = threadIdx.x;
  const int w    = tid >> 5;
  const int lane = tid & 31;
  const int lr   = lane & 15;   // N-col / row-within-half for fragments
  const int hi   = lane >> 4;   // half-wave select
  const int n0   = blockIdx.x * 128 + w * 16;

  // per-thread copy bases (hoisted out of the stage loops)
  const unsigned int* gbase = wsall + tid * 16;
  const unsigned lb0 = (unsigned)(uintptr_t)(void*)s_stage + (unsigned)tid * 64u;
  const unsigned lb1 = lb0 + 16384u;

  // gates for this WG's 128 tokens
  s_g[tid * 2 + 0] = gates[(size_t)blockIdx.x * 512 + tid * 2 + 0];
  s_g[tid * 2 + 1] = gates[(size_t)blockIdx.x * 512 + tid * 2 + 1];

  const v8f vzero = {0.f, 0.f, 0.f, 0.f, 0.f, 0.f, 0.f, 0.f};
  v8f yacc[16];                       // y tile [16 x 256] f32 accumulators
#pragma unroll
  for (int j = 0; j < 16; ++j) yacc[j] = vzero;

  // ---- prologue: async-copy w1 chunk 0 into buffer 0 ----
  stage_prefetch(gbase, lb0, 0u);
  wait_async0();

  // ======================= loop A: experts (GEMM1+GEMM2) ===================
  {
    // x A-fragments (16x256 -> 8 K-chunks of 32), resident only in this scope.
    // A layout: lanes 0-15 row=lr K=c*32+0..15; 16-31 K=c*32+16..31
    v16h xf[8];
    {
      const float* xr = x + (size_t)(n0 + lr) * D_IN;
#pragma unroll
      for (int c = 0; c < 8; ++c) {
        const int d0 = c * 32 + hi * 16;
        f32x4 q0 = *(const f32x4*)(xr + d0);
        f32x4 q1 = *(const f32x4*)(xr + d0 + 4);
        f32x4 q2 = *(const f32x4*)(xr + d0 + 8);
        f32x4 q3 = *(const f32x4*)(xr + d0 + 12);
        v16h f;
#pragma unroll
        for (int i = 0; i < 4; ++i) {
          f[i]      = (_Float16)q0[i];
          f[4 + i]  = (_Float16)q1[i];
          f[8 + i]  = (_Float16)q2[i];
          f[12 + i] = (_Float16)q3[i];
        }
        xf[c] = f;
      }
    }

    v16h hfrag = {};   // GEMM1 -> GEMM2 handoff (per K-chunk)

#pragma unroll 1
    for (int k = 0; k < 64; ++k) {       // (e, c2) = (k>>4, k&15)
      const int e = k >> 4, c2 = k & 15;

      // ---------- phase 1: GEMM1 from buf0; prefetch w2(e,c2) -> buf1 ------
      __syncthreads();
      stage_prefetch(gbase, lb1, 262144u + (unsigned)(e * 16 + c2) * 4096u);
#pragma unroll
      for (int jj = 0; jj < 2; ++jj) {   // two 16-col h tiles of K-chunk c2
        v8f acc = vzero;
#pragma unroll
        for (int c = 0; c < 8; ++c)
          acc = wmma16(xf[c],
                       frag_from_dwords(s_stage + (jj * 8 + c) * 256 + lane * 8), acc);
        // C layout: VGPR v -> row m = v + hi*8, col = lr
#pragma unroll
        for (int v = 0; v < 8; ++v) {
          const int m = v + hi * 8;
          const float f  = acc[v];
          const float gl = 0.5f * f * (1.0f + erff(f * 0.70710678118654752f));
          const float g  = s_g[(w * 16 + m) * 4 + e];
          s_hb[w][m][jj * 16 + lr] = (_Float16)(g * gl);  // same-wave LDS: in order
        }
      }
      hfrag = frag_from_halves(&s_hb[w][lr][hi * 16], &s_hb[w][lr][hi * 16 + 8]);
      wait_async0();

      // ---------- phase 2: GEMM2 from buf1; prefetch next w1 -> buf0 -------
      __syncthreads();
      {
        const int kn = k + 1;
        const unsigned off = (kn < 64)
            ? (unsigned)((kn >> 4) * 65536 + (kn & 15) * 4096)
            : 524288u;                      // first w_proj chunk
        stage_prefetch(gbase, lb0, off);
      }
#pragma unroll
      for (int j2 = 0; j2 < 16; ++j2)
        yacc[j2] = wmma16(hfrag,
                          frag_from_dwords(s_stage + 4096 + j2 * 256 + lane * 8),
                          yacc[j2]);
      wait_async0();
    }
  }

  // ---- build y A-fragments via per-wave bounce (yacc dies here) ----
  v16h yf[8];
#pragma unroll
  for (int c3 = 0; c3 < 8; ++c3) {
#pragma unroll
    for (int t = 0; t < 2; ++t) {
#pragma unroll
      for (int v = 0; v < 8; ++v) {
        const int m = v + hi * 8;
        s_hb[w][m][t * 16 + lr] = (_Float16)yacc[c3 * 2 + t][v];
      }
    }
    yf[c3] = frag_from_halves(&s_hb[w][lr][hi * 16], &s_hb[w][lr][hi * 16 + 8]);
  }

  // ======================= loop B: GEMM3 (+bias) ===========================
#pragma unroll 1
  for (int t = 0; t < 16; ++t) {
    __syncthreads();
    if (t + 1 < 16)
      stage_prefetch(gbase, ((t + 1) & 1) ? lb1 : lb0,
                     524288u + (unsigned)(t + 1) * 4096u);
    const unsigned int* stg = s_stage + (t & 1) * 4096;
#pragma unroll
    for (int jj = 0; jj < 2; ++jj) {
      const int j3 = t * 2 + jj;
      v8f acc = vzero;
#pragma unroll
      for (int c3 = 0; c3 < 8; ++c3)
        acc = wmma16(yf[c3],
                     frag_from_dwords(stg + (jj * 8 + c3) * 256 + lane * 8), acc);
      const float bb = b_proj[j3 * 16 + lr];
#pragma unroll
      for (int v = 0; v < 8; ++v) {
        const int m = v + hi * 8;
        out[(size_t)(n0 + m) * EMB_DIM + j3 * 16 + lr] = acc[v] + bb;
      }
    }
    wait_async0();
  }
}

// ---------------------------------------------------------------------------
extern "C" void kernel_launch(void* const* d_in, const int* in_sizes, int n_in,
                              void* d_out, int out_size, void* d_ws, size_t ws_size,
                              hipStream_t stream)
{
  const float* x      = (const float*)d_in[0];   // [N, 256]
  const float* w_gate = (const float*)d_in[1];   // [256, 4]
  const float* w1     = (const float*)d_in[2];   // [4, 256, 512]
  const float* w2     = (const float*)d_in[3];   // [4, 512, 256]
  const float* w_proj = (const float*)d_in[4];   // [256, 512]
  const float* b_proj = (const float*)d_in[5];   // [512]
  float* out = (float*)d_out;

  const int N = in_sizes[0] / D_IN;              // 65536

  unsigned int* ws  = (unsigned int*)d_ws;
  unsigned int* w1p = ws;                        // 262144 dwords
  unsigned int* w2p = ws + 262144;               // 262144 dwords
  unsigned int* wpp = ws + 524288;               //  65536 dwords
  float* gates      = (float*)(ws + 589824);     // 262144 floats

  // pack weights to f16 fragment order
  pack_w_kernel<<<262144 / 256, 256, 0, stream>>>(w1, w1p, 32, 8, 512, 0, 262144);
  pack_w_kernel<<<262144 / 256, 256, 0, stream>>>(w2, w2p, 16, 16, 256, 1, 262144);
  pack_w_kernel<<<65536 / 256, 256, 0, stream>>>(w_proj, wpp, 32, 8, 512, 0, 65536);
  // dense top-2 softmax gate matrix
  gate_topk_kernel<<<N / 8, 256, 0, stream>>>(x, w_gate, gates);
  // fused MoE + projection (single packed-weights base pointer)
  moe_fused_kernel<<<N / 128, WG_THREADS, 0, stream>>>(x, b_proj, ws, gates, out);
}
